// ConditionedTransformerPairBiasLayer_22780506538106
// MI455X (gfx1250) — matrix operand order repaired and checked by reading, not compiled, not run on hardware
//
#include <hip/hip_runtime.h>

// ---------------------------------------------------------------------------
// ConditionedTransformerPairBiasLayer for MI455X (gfx1250, wave32, WMMA+TDM).
// B=2, N=512, C_S=1024, C_COND=512, C_Z=128, H=16, D=64.
// Dense math: v_wmma_f32_16x16x32_bf16. A-tile staging via Tensor Data Mover
// (tensor_load_to_lds + s_wait_tensorcnt), double-buffered so the TDM DMA
// overlaps the WMMA pipe; B tiles double-buffered through registers.
// ---------------------------------------------------------------------------

typedef __bf16 bf16_t;
typedef __attribute__((ext_vector_type(16))) __bf16 bf16x16;
typedef __attribute__((ext_vector_type(8)))  float  floatx8;
typedef unsigned int uint32x4 __attribute__((ext_vector_type(4)));
typedef int int32x8 __attribute__((ext_vector_type(8)));
typedef int int32x4 __attribute__((ext_vector_type(4)));

#define TM 128
#define TN 64
#define TK 32
#define LDP 40   // padded LDS row pitch in halves (80B, 16B-aligned)

// ---- TDM dispatch: header => 6-arg builtin; else bare builtin => 5-arg form.
#if __has_include(<hip/amd_detail/amd_gfx1250_TDM.h>)
  #define TDM_ARITY 6
#elif __has_builtin(__builtin_amdgcn_tensor_load_to_lds)
  #define TDM_ARITY 5
#else
  #define TDM_ARITY 0
#endif

enum { Bc = 2, Nn = 512, CS = 1024, CC = 512, CZ = 128, Hh = 16, Dd = 64, BN = Bc * Nn };

struct GemmParams {
    const bf16_t* A;   long long lda, sAb, sAh;        // A[m,k], k contiguous
    const bf16_t* Bm;  long long ldbk, ldbn, sBb, sBh; // B elem = base + k*ldbk + n*ldbn
    float*  Cf;  bf16_t* Cb;
    long long ldc, sCb, sCh;                           // C elem = base + m*ldc + n
    int M, N, K, H;                                    // grid.z batches, bh = z % H
    float alpha;
    const float* pb;                                   // [b, q, k, H] pair bias (epi 2)
    const float* mask;                                 // [b, N] (epi 2)
};

__device__ __forceinline__ float warp_sum(float v) {
    #pragma unroll
    for (int o = 16; o > 0; o >>= 1) v += __shfl_xor(v, o, 32);
    return v;
}
__device__ __forceinline__ float warp_max(float v) {
    #pragma unroll
    for (int o = 16; o > 0; o >>= 1) v = fmaxf(v, __shfl_xor(v, o, 32));
    return v;
}
__device__ __forceinline__ float sigf(float x) { return 1.0f / (1.0f + expf(-x)); }

#if TDM_ARITY > 0
// Issue one TDM 2D tile load: rows x 32 halves, global(k-contig, stride lda)
// -> LDS at lds_off with 16B padding after every 64B row (pitch 80B = LDP).
__device__ __forceinline__ void tdm_load_tile(const bf16_t* gsrc, unsigned lds_off,
                                              long long lda_elems, int rows) {
    unsigned long long ga = (unsigned long long)(size_t)gsrc;
    uint32x4 g0;
    g0.x = 1u;                                              // count=1, user mode
    g0.y = lds_off;                                         // LDS byte address
    g0.z = (unsigned)(ga & 0xFFFFFFFFu);                    // global_addr[31:0]
    g0.w = (unsigned)((ga >> 32) & 0x1FFFFFFu) | (2u << 30);// addr[56:32] | type=2
    unsigned long long s0 = (unsigned long long)lda_elems;  // dim0 stride (elems)
    unsigned td0 = 0x40000000u, td1 = 0x40000000u;          // huge dims (no OOB clip)
    int32x8 g1;
    g1[0] = (int)((1u << 16)   /* data_size = 2B   */
                | (1u << 20)   /* pad_enable       */
                | (3u << 22)   /* pad every 16 dw  */
                | (3u << 25)); /* pad amount 4 dw  */
    g1[1] = (int)((td0 & 0xFFFFu) << 16);
    g1[2] = (int)(((td0 >> 16) & 0xFFFFu) | ((td1 & 0xFFFFu) << 16));
    g1[3] = (int)(((td1 >> 16) & 0xFFFFu) | ((unsigned)TK << 16)); // tile_dim0=32
    g1[4] = (int)((unsigned)rows);                                  // tile_dim1
    g1[5] = (int)(unsigned)(s0 & 0xFFFFFFFFu);
    g1[6] = (int)(unsigned)((s0 >> 32) & 0xFFFFu);                  // stride1=0
    g1[7] = 0;
    int32x4 zz = {0, 0, 0, 0};
#if TDM_ARITY == 5
    __builtin_amdgcn_tensor_load_to_lds(g0, g1, zz, zz, 0);
#else
    int32x8 zz8 = {0, 0, 0, 0, 0, 0, 0, 0};
    __builtin_amdgcn_tensor_load_to_lds(g0, g1, zz, zz, zz8, 0);
#endif
}
#endif

// ---- B tile helpers: load one 16B chunk / store (transposing) into Bs[n][k]
template<int BMODE>
__device__ __forceinline__ uint4 loadB(const bf16_t* Bb, const GemmParams& p,
                                       int n0, int k0, int tid) {
    if constexpr (BMODE == 0) {         // contiguous along k
        int nb = tid >> 2, kc = (tid & 3) * 8;
        return *(const uint4*)(Bb + (long long)(n0 + nb) * p.ldbn + (k0 + kc));
    } else {                            // contiguous along n
        int k = tid >> 3, nc = (tid & 7) * 8;
        return *(const uint4*)(Bb + (long long)(k0 + k) * p.ldbk + (n0 + nc));
    }
}
template<int BMODE>
__device__ __forceinline__ void storeB(bf16_t* bs, uint4 d, int tid) {
    if constexpr (BMODE == 0) {
        int nb = tid >> 2, kc = (tid & 3) * 8;
        *(uint4*)&bs[nb * LDP + kc] = d;
    } else {
        int k = tid >> 3, nc = (tid & 7) * 8;
        const bf16_t* h = (const bf16_t*)&d;
        #pragma unroll
        for (int i = 0; i < 8; ++i) bs[(nc + i) * LDP + k] = h[i];
    }
}

#if TDM_ARITY == 0
__device__ __forceinline__ void loadA(uint4* r, const bf16_t* Ab, const GemmParams& p,
                                      int m0, int k0, int tid) {
    #pragma unroll
    for (int c = 0; c < 2; ++c) {
        int chunk = tid + c * 256;
        int row = chunk >> 2, kc = (chunk & 3) * 8;
        r[c] = *(const uint4*)(Ab + (long long)(m0 + row) * p.lda + (k0 + kc));
    }
}
__device__ __forceinline__ void storeA(bf16_t* as, const uint4* r, int tid) {
    #pragma unroll
    for (int c = 0; c < 2; ++c) {
        int chunk = tid + c * 256;
        int row = chunk >> 2, kc = (chunk & 3) * 8;
        *(uint4*)&as[row * LDP + kc] = r[c];
    }
}
#endif

// ---------------------------------------------------------------------------
// Generic batched strided WMMA GEMM: C = A @ B (+ epilogue)
// Block = 256 threads (8 waves), 128x64 C tile, 32x32 per wave.
// Double-buffered: TDM (A) and register-staged global loads (B) for tile k+1
// are issued before the WMMAs of tile k; waits/stores land after the math.
// BMODE 0: B contiguous along k. BMODE 1: contiguous along n.
// EPI   0: f32 store. 1: bf16 store. 2: scores epilogue (scale+bias+mask).
// ---------------------------------------------------------------------------
template<int BMODE, int EPI>
__global__ __launch_bounds__(256) void wmma_gemm_kernel(GemmParams p) {
    __shared__ bf16_t As[2][TM * LDP];       // As[buf][m][k]
    __shared__ bf16_t Bs[2][TN * LDP];       // Bs[buf][n][k] (transposed stage)

    const int bz = blockIdx.z;
    const int bh = bz % p.H;
    const int bb = bz / p.H;
    const bf16_t* Ab = p.A  + (long long)bb * p.sAb + (long long)bh * p.sAh;
    const bf16_t* Bb = p.Bm + (long long)bb * p.sBb + (long long)bh * p.sBh;

    const int m0 = blockIdx.y * TM;
    const int n0 = blockIdx.x * TN;
    const int tid  = threadIdx.x;
    const int lane = tid & 31;
    const int wid  = tid >> 5;            // 0..7
    const int wm   = (wid >> 1) * 32;     // 0,32,64,96
    const int wn   = (wid & 1) * 32;      // 0,32
    const int half = lane >> 4;
    const int l15  = lane & 15;

    floatx8 acc[2][2];
    #pragma unroll
    for (int i = 0; i < 2; ++i)
        #pragma unroll
        for (int j = 0; j < 2; ++j) { floatx8 zv = {}; acc[i][j] = zv; }

    // ---- prologue: stage tile 0 into buffer 0
#if TDM_ARITY > 0
    if (wid == 0)
        tdm_load_tile(Ab + (long long)m0 * p.lda, (unsigned)(size_t)&As[0][0], p.lda, TM);
#else
    { uint4 ar[2]; loadA(ar, Ab, p, m0, 0, tid); storeA(&As[0][0], ar, tid); }
#endif
    { uint4 br = loadB<BMODE>(Bb, p, n0, 0, tid); storeB<BMODE>(&Bs[0][0], br, tid); }
#if TDM_ARITY > 0
    if (wid == 0) __builtin_amdgcn_s_wait_tensorcnt(0);
#endif
    __syncthreads();

    int buf = 0;
    for (int k0 = 0; k0 < p.K; k0 += TK) {
        const int k1 = k0 + TK;
        const bool more = k1 < p.K;

        // ---- kick off tile k+1 (DMA for A, register loads for B)
        uint4 brNext = {};
#if TDM_ARITY == 0
        uint4 arNext[2] = {};
#endif
        if (more) {
            if (k1 + TK < p.K) {     // prefetch two tiles ahead (global_prefetch_b8)
                __builtin_prefetch(Ab + (long long)(m0 + (tid >> 1)) * p.lda + (k1 + TK), 0, 1);
                __builtin_prefetch(Bb + (long long)(k1 + TK) * p.ldbk
                                      + (long long)(n0 + (tid & 63)) * p.ldbn, 0, 1);
            }
#if TDM_ARITY > 0
            if (wid == 0)
                tdm_load_tile(Ab + (long long)m0 * p.lda + k1,
                              (unsigned)(size_t)&As[buf ^ 1][0], p.lda, TM);
#else
            loadA(arNext, Ab, p, m0, k1, tid);
#endif
            brNext = loadB<BMODE>(Bb, p, n0, k1, tid);
        }

        // ---- compute on buffer `buf`
        union FragU { bf16x16 v; uint4 u[2]; };
        bf16x16 afr[2], bfr[2];
        #pragma unroll
        for (int i = 0; i < 2; ++i) {
            int m = wm + i * 16 + l15;
            int kb = half * 8;
            FragU f;
            f.u[0] = *(uint4*)&As[buf][m * LDP + kb];
            f.u[1] = *(uint4*)&As[buf][m * LDP + 16 + kb];
            afr[i] = f.v;
        }
        #pragma unroll
        for (int j = 0; j < 2; ++j) {
            int n = wn + j * 16 + l15;
            int kb = half * 8;
            FragU f;
            f.u[0] = *(uint4*)&Bs[buf][n * LDP + kb];
            f.u[1] = *(uint4*)&Bs[buf][n * LDP + kb + 16];
            bfr[j] = f.v;
        }
        #pragma unroll
        for (int i = 0; i < 2; ++i)
            #pragma unroll
            for (int j = 0; j < 2; ++j)
                acc[i][j] = __builtin_amdgcn_wmma_f32_16x16x32_bf16(
                    false, afr[i], false, bfr[j], (short)0, acc[i][j], false, false);

        // ---- land tile k+1 into the other buffer
        if (more) {
            storeB<BMODE>(&Bs[buf ^ 1][0], brNext, tid);
#if TDM_ARITY == 0
            storeA(&As[buf ^ 1][0], arNext, tid);
#else
            if (wid == 0) __builtin_amdgcn_s_wait_tensorcnt(0);
#endif
        }
        __syncthreads();
        buf ^= 1;
    }

    // ---- epilogue / store
    #pragma unroll
    for (int i = 0; i < 2; ++i) {
        #pragma unroll
        for (int j = 0; j < 2; ++j) {
            int n = n0 + wn + j * 16 + l15;
            #pragma unroll
            for (int r = 0; r < 8; ++r) {
                int m = m0 + wm + i * 16 + half * 8 + r;
                float v = acc[i][j][r];
                if constexpr (EPI == 2) {
                    float pbv = p.pb[(((long long)bb * p.M + m) * p.N + n) * p.H + bh];
                    float mk = p.mask[(long long)bb * p.N + n];
                    float mq = p.mask[(long long)bb * p.M + m];
                    v = v * p.alpha + pbv + 1.0e8f * (mk - 1.0f) + 1.0e8f * (mq - 1.0f);
                }
                long long ci = (long long)bb * p.sCb + (long long)bh * p.sCh
                             + (long long)m * p.ldc + n;
                if constexpr (EPI == 1) p.Cb[ci] = (bf16_t)v;
                else                    p.Cf[ci] = v;
            }
        }
    }
}

// ---------------------------------------------------------------------------
// f32 -> bf16 convert
// ---------------------------------------------------------------------------
__global__ __launch_bounds__(256) void f2bf_kernel(const float* __restrict__ s,
                                                   bf16_t* __restrict__ d, long long n) {
    long long i = (long long)blockIdx.x * 256 + threadIdx.x;
    if (i < n) d[i] = (bf16_t)s[i];
}

// ---------------------------------------------------------------------------
// Row LayerNorm: one block per row; optional affine; bf16 and/or f32 out
// ---------------------------------------------------------------------------
__global__ __launch_bounds__(256) void ln_kernel(const float* __restrict__ X,
                                                 const float* __restrict__ sc,
                                                 const float* __restrict__ bi,
                                                 bf16_t* __restrict__ outb,
                                                 float* __restrict__ outf, int C) {
    long long row = blockIdx.x;
    const float* xr = X + row * C;
    int tid = threadIdx.x, lane = tid & 31, wid = tid >> 5;
    float s = 0.f, q = 0.f;
    for (int c = tid; c < C; c += 256) { float v = xr[c]; s += v; q += v * v; }
    s = warp_sum(s); q = warp_sum(q);
    __shared__ float ss[8], qq[8];
    if (!lane) { ss[wid] = s; qq[wid] = q; }
    __syncthreads();
    float ts = 0.f, tq = 0.f;
    #pragma unroll
    for (int i = 0; i < 8; ++i) { ts += ss[i]; tq += qq[i]; }
    float mean = ts / C;
    float var  = tq / C - mean * mean;
    float inv  = rsqrtf(var + 1e-5f);
    for (int c = tid; c < C; c += 256) {
        float v = (xr[c] - mean) * inv;
        if (sc) v = v * sc[c] + bi[c];
        if (outb) outb[row * C + c] = (bf16_t)v;
        if (outf) outf[row * C + c] = v;
    }
}

// ---------------------------------------------------------------------------
// Pair bias: one wave per (b,i,j) row of z. LN over 128 ch + 128x16 matvec.
// pb layout: [b, i, j, H]
// ---------------------------------------------------------------------------
__global__ __launch_bounds__(256) void pairbias_kernel(const float* __restrict__ z,
                                                       const float* __restrict__ zs,
                                                       const float* __restrict__ zb,
                                                       const float* __restrict__ wb,
                                                       float* __restrict__ pb) {
    __shared__ float wbs[CZ * Hh];
    __shared__ float zss[CZ], zbs[CZ];
    for (int i = threadIdx.x; i < CZ * Hh; i += 256) wbs[i] = wb[i];
    for (int i = threadIdx.x; i < CZ; i += 256) { zss[i] = zs[i]; zbs[i] = zb[i]; }
    __syncthreads();

    long long row = (long long)blockIdx.x * 8 + (threadIdx.x >> 5);
    int lane = threadIdx.x & 31;
    const float4 v4 = *(const float4*)(z + row * CZ + lane * 4);
    float v[4] = { v4.x, v4.y, v4.z, v4.w };
    float s = v[0] + v[1] + v[2] + v[3];
    float q = v[0]*v[0] + v[1]*v[1] + v[2]*v[2] + v[3]*v[3];
    s = warp_sum(s); q = warp_sum(q);
    float mean = s * (1.0f / CZ);
    float var  = q * (1.0f / CZ) - mean * mean;
    float inv  = rsqrtf(var + 1e-5f);
    float nz[4];
    #pragma unroll
    for (int j = 0; j < 4; ++j)
        nz[j] = (v[j] - mean) * inv * zss[lane * 4 + j] + zbs[lane * 4 + j];
    #pragma unroll
    for (int h = 0; h < Hh; ++h) {
        float p = 0.f;
        #pragma unroll
        for (int j = 0; j < 4; ++j) p += nz[j] * wbs[(lane * 4 + j) * Hh + h];
        p = warp_sum(p);
        if (lane == h) pb[row * Hh + h] = p;
    }
}

// ---------------------------------------------------------------------------
// Row softmax (C=512), writes bf16 attn
// ---------------------------------------------------------------------------
__global__ __launch_bounds__(256) void softmax_kernel(const float* __restrict__ S,
                                                      bf16_t* __restrict__ A, int C) {
    long long row = blockIdx.x;
    const float* sr = S + row * C;
    int tid = threadIdx.x, lane = tid & 31, wid = tid >> 5;
    float m = -3.0e38f;
    for (int c = tid; c < C; c += 256) m = fmaxf(m, sr[c]);
    m = warp_max(m);
    __shared__ float shm[8], shs[8];
    if (!lane) shm[wid] = m;
    __syncthreads();
    m = shm[0];
    #pragma unroll
    for (int i = 1; i < 8; ++i) m = fmaxf(m, shm[i]);
    float s = 0.f;
    for (int c = tid; c < C; c += 256) s += expf(sr[c] - m);
    s = warp_sum(s);
    if (!lane) shs[wid] = s;
    __syncthreads();
    float tot = 0.f;
    #pragma unroll
    for (int i = 0; i < 8; ++i) tot += shs[i];
    float invs = 1.0f / tot;
    for (int c = tid; c < C; c += 256)
        A[row * C + c] = (bf16_t)(expf(sr[c] - m) * invs);
}

// ---------------------------------------------------------------------------
// Fused elementwise kernels
// ---------------------------------------------------------------------------
__global__ __launch_bounds__(256) void modulate_kernel(const float* __restrict__ xln,
                                                       const float* __restrict__ G,
                                                       const float* __restrict__ gb,
                                                       const float* __restrict__ Bt,
                                                       bf16_t* __restrict__ xmod, long long n) {
    long long i = (long long)blockIdx.x * 256 + threadIdx.x;
    if (i < n) {
        int c = (int)(i & (CS - 1));
        xmod[i] = (bf16_t)(xln[i] * sigf(G[i] + gb[c]) + Bt[i]);
    }
}

__global__ __launch_bounds__(256) void gate_kernel(const float* __restrict__ upd,
                                                   const float* __restrict__ og,
                                                   bf16_t* __restrict__ updg, long long n) {
    long long i = (long long)blockIdx.x * 256 + threadIdx.x;
    if (i < n) updg[i] = (bf16_t)(upd[i] * sigf(og[i]));
}

__global__ __launch_bounds__(256) void cgate_res_kernel(const float* __restrict__ x,
                                                        const float* __restrict__ o2,
                                                        const float* __restrict__ cg,
                                                        const float* __restrict__ bcg,
                                                        float* __restrict__ xa, long long n) {
    long long i = (long long)blockIdx.x * 256 + threadIdx.x;
    if (i < n) {
        int c = (int)(i & (CS - 1));
        xa[i] = x[i] + o2[i] * sigf(cg[i] + bcg[c]);
    }
}

__global__ __launch_bounds__(256) void swiglu_kernel(const float* __restrict__ a1,
                                                     const float* __restrict__ a2,
                                                     bf16_t* __restrict__ hg, long long n) {
    long long i = (long long)blockIdx.x * 256 + threadIdx.x;
    if (i < n) {
        float a = a1[i];
        hg[i] = (bf16_t)(a * sigf(a) * a2[i]);
    }
}

__global__ __launch_bounds__(256) void final_kernel(const float* __restrict__ xa,
                                                    const float* __restrict__ F,
                                                    const float* __restrict__ mask,
                                                    float* __restrict__ out, long long n) {
    long long i = (long long)blockIdx.x * 256 + threadIdx.x;
    if (i < n) out[i] = xa[i] + F[i] * mask[i >> 10];   // CS == 1024
}

// ---------------------------------------------------------------------------
// Host launch helpers
// ---------------------------------------------------------------------------
static void launch_gemm(hipStream_t st,
                        const bf16_t* A, long long lda, long long sAb, long long sAh,
                        const bf16_t* Bm, long long ldbk, long long ldbn, long long sBb, long long sBh,
                        float* Cf, bf16_t* Cb, long long ldc, long long sCb, long long sCh,
                        int M, int N, int K, int Hheads, int batches,
                        int epi, float alpha, const float* pb, const float* mask) {
    GemmParams p;
    p.A = A; p.lda = lda; p.sAb = sAb; p.sAh = sAh;
    p.Bm = Bm; p.ldbk = ldbk; p.ldbn = ldbn; p.sBb = sBb; p.sBh = sBh;
    p.Cf = Cf; p.Cb = Cb; p.ldc = ldc; p.sCb = sCb; p.sCh = sCh;
    p.M = M; p.N = N; p.K = K; p.H = Hheads;
    p.alpha = alpha; p.pb = pb; p.mask = mask;
    dim3 grid(N / TN, M / TM, batches);
    dim3 blk(256);
    // exactly the three configurations used by this layer
    if (epi == 2)      wmma_gemm_kernel<0, 2><<<grid, blk, 0, st>>>(p);  // scores (B k-contig)
    else if (epi == 1) wmma_gemm_kernel<1, 1><<<grid, blk, 0, st>>>(p);  // bf16 out (weights)
    else               wmma_gemm_kernel<1, 0><<<grid, blk, 0, st>>>(p);  // f32 out (weights/V)
}

static void conv_bf16(hipStream_t st, const float* s, bf16_t* d, long long n) {
    f2bf_kernel<<<(int)((n + 255) / 256), 256, 0, st>>>(s, d, n);
}

extern "C" void kernel_launch(void* const* d_in, const int* in_sizes, int n_in,
                              void* d_out, int out_size, void* d_ws, size_t ws_size,
                              hipStream_t stream) {
    const float* x      = (const float*)d_in[0];
    const float* cond   = (const float*)d_in[1];
    const float* z      = (const float*)d_in[2];
    const float* x_mask = (const float*)d_in[3];
    const float* gamma_w= (const float*)d_in[4];
    const float* gamma_b= (const float*)d_in[5];
    const float* beta_w = (const float*)d_in[6];
    const float* z_scale= (const float*)d_in[7];
    const float* z_bias = (const float*)d_in[8];
    const float* w_q    = (const float*)d_in[9];
    const float* w_kv   = (const float*)d_in[10];
    const float* w_b    = (const float*)d_in[11];
    const float* w_og   = (const float*)d_in[12];
    const float* w_out  = (const float*)d_in[13];
    const float* w_cg   = (const float*)d_in[14];
    const float* b_cg   = (const float*)d_in[15];
    const float* ffn_s  = (const float*)d_in[16];
    const float* ffn_b  = (const float*)d_in[17];
    const float* w_a    = (const float*)d_in[18];
    const float* w_b2   = (const float*)d_in[19];
    const float* w_o    = (const float*)d_in[20];
    float* out = (float*)d_out;

    // ---- workspace bump allocator (256B aligned)
    char* wp = (char*)d_ws;
    auto alloc = [&](size_t bytes) -> void* {
        void* r = wp; wp += (bytes + 255) & ~(size_t)255; return r;
    };
    bf16_t* g_gamma = (bf16_t*)alloc((size_t)CC * CS * 2);
    bf16_t* g_beta  = (bf16_t*)alloc((size_t)CC * CS * 2);
    bf16_t* g_wq    = (bf16_t*)alloc((size_t)CS * CS * 2);
    bf16_t* g_wkv   = (bf16_t*)alloc((size_t)CS * 2 * CS * 2);
    bf16_t* g_wog   = (bf16_t*)alloc((size_t)CS * CS * 2);
    bf16_t* g_wout  = (bf16_t*)alloc((size_t)CS * CS * 2);
    bf16_t* g_wcg   = (bf16_t*)alloc((size_t)CC * CS * 2);
    bf16_t* g_wa    = (bf16_t*)alloc((size_t)CS * 2 * CS * 2);
    bf16_t* g_wb2   = (bf16_t*)alloc((size_t)CS * 2 * CS * 2);
    bf16_t* g_wo    = (bf16_t*)alloc((size_t)2 * CS * CS * 2);
    bf16_t* condbf  = (bf16_t*)alloc((size_t)BN * CC * 2);
    bf16_t* condn   = (bf16_t*)alloc((size_t)BN * CC * 2);
    float*  xln     = (float*)alloc((size_t)BN * CS * 4);
    float*  Gm      = (float*)alloc((size_t)BN * CS * 4);
    float*  Bt      = (float*)alloc((size_t)BN * CS * 4);
    float*  CGl     = (float*)alloc((size_t)BN * CS * 4);
    bf16_t* xmod    = (bf16_t*)alloc((size_t)BN * CS * 2);
    bf16_t* Qb      = (bf16_t*)alloc((size_t)BN * CS * 2);
    bf16_t* KVb     = (bf16_t*)alloc((size_t)BN * 2 * CS * 2);
    float*  OGf     = (float*)alloc((size_t)BN * CS * 4);
    float*  pbuf    = (float*)alloc((size_t)Bc * Nn * Nn * Hh * 4);
    float*  scores  = (float*)alloc((size_t)Bc * Hh * Nn * Nn * 4);
    bf16_t* attnb   = (bf16_t*)alloc((size_t)Bc * Hh * Nn * Nn * 2);
    float*  upd     = (float*)alloc((size_t)BN * CS * 4);
    bf16_t* updg    = (bf16_t*)alloc((size_t)BN * CS * 2);
    float*  out2    = (float*)alloc((size_t)BN * CS * 4);
    float*  xa      = (float*)alloc((size_t)BN * CS * 4);
    bf16_t* xlb     = (bf16_t*)alloc((size_t)BN * CS * 2);
    float*  A1      = (float*)alloc((size_t)BN * 2 * CS * 4);
    float*  A2      = (float*)alloc((size_t)BN * 2 * CS * 4);
    bf16_t* hg      = (bf16_t*)alloc((size_t)BN * 2 * CS * 2);
    float*  Ff      = (float*)alloc((size_t)BN * CS * 4);

    // ---- weight / input bf16 conversion
    conv_bf16(stream, gamma_w, g_gamma, (long long)CC * CS);
    conv_bf16(stream, beta_w,  g_beta,  (long long)CC * CS);
    conv_bf16(stream, w_q,     g_wq,    (long long)CS * CS);
    conv_bf16(stream, w_kv,    g_wkv,   (long long)CS * 2 * CS);
    conv_bf16(stream, w_og,    g_wog,   (long long)CS * CS);
    conv_bf16(stream, w_out,   g_wout,  (long long)CS * CS);
    conv_bf16(stream, w_cg,    g_wcg,   (long long)CC * CS);
    conv_bf16(stream, w_a,     g_wa,    (long long)CS * 2 * CS);
    conv_bf16(stream, w_b2,    g_wb2,   (long long)CS * 2 * CS);
    conv_bf16(stream, w_o,     g_wo,    (long long)2 * CS * CS);
    conv_bf16(stream, cond,    condbf,  (long long)BN * CC);

    // ---- LN(cond) -> bf16 ; LN(x) -> f32
    ln_kernel<<<BN, 256, 0, stream>>>(cond, nullptr, nullptr, condn, nullptr, CC);
    ln_kernel<<<BN, 256, 0, stream>>>(x,    nullptr, nullptr, nullptr, xln,   CS);

    // ---- AdaLN projections: G = condn @ gamma_w ; Bt = condn @ beta_w ; CG = cond @ w_cg
    launch_gemm(stream, condn, CC, 0, 0, g_gamma, CS, 1, 0, 0,
                Gm, nullptr, CS, 0, 0, BN, CS, CC, 1, 1, 0, 1.0f, nullptr, nullptr);
    launch_gemm(stream, condn, CC, 0, 0, g_beta, CS, 1, 0, 0,
                Bt, nullptr, CS, 0, 0, BN, CS, CC, 1, 1, 0, 1.0f, nullptr, nullptr);
    launch_gemm(stream, condbf, CC, 0, 0, g_wcg, CS, 1, 0, 0,
                CGl, nullptr, CS, 0, 0, BN, CS, CC, 1, 1, 0, 1.0f, nullptr, nullptr);

    // ---- modulated input
    {
        long long n = (long long)BN * CS;
        modulate_kernel<<<(int)((n + 255) / 256), 256, 0, stream>>>(xln, Gm, gamma_b, Bt, xmod, n);
    }

    // ---- Q / KV / OG projections (Q, KV stored bf16 for attention GEMMs)
    launch_gemm(stream, xmod, CS, 0, 0, g_wq, CS, 1, 0, 0,
                nullptr, Qb, CS, 0, 0, BN, CS, CS, 1, 1, 1, 1.0f, nullptr, nullptr);
    launch_gemm(stream, xmod, CS, 0, 0, g_wkv, 2 * CS, 1, 0, 0,
                nullptr, KVb, 2 * CS, 0, 0, BN, 2 * CS, CS, 1, 1, 1, 1.0f, nullptr, nullptr);
    launch_gemm(stream, xmod, CS, 0, 0, g_wog, CS, 1, 0, 0,
                OGf, nullptr, CS, 0, 0, BN, CS, CS, 1, 1, 0, 1.0f, nullptr, nullptr);

    // ---- pair bias from z (the 268MB streaming pass)
    {
        long long rows = (long long)Bc * Nn * Nn;
        pairbias_kernel<<<(int)(rows / 8), 256, 0, stream>>>(z, z_scale, z_bias, w_b, pbuf);
    }

    // ---- scores = Q @ K^T * scale + bias + mask   (batched over B*H)
    launch_gemm(stream,
                Qb, CS, (long long)Nn * CS, Dd,                         // A: Q[b, m, h*64 + k]
                KVb, 1, 2 * CS, (long long)Nn * 2 * CS, Dd,             // B: K[b, n, h*64 + k]
                scores, nullptr, Nn, (long long)Hh * Nn * Nn, (long long)Nn * Nn,
                Nn, Nn, Dd, Hh, Bc * Hh, 2, 0.125f, pbuf, x_mask);

    // ---- softmax -> bf16 attn
    softmax_kernel<<<Bc * Hh * Nn, 256, 0, stream>>>(scores, attnb, Nn);

    // ---- update = attn @ V  (batched over B*H), scattered into [b, q, h*64+d]
    launch_gemm(stream,
                attnb, Nn, (long long)Hh * Nn * Nn, (long long)Nn * Nn, // A: attn[b,h,q,k]
                KVb + CS, 2 * CS, 1, (long long)Nn * 2 * CS, Dd,        // B: V[b, k, h*64 + d]
                upd, nullptr, CS, (long long)Nn * CS, Dd,
                Nn, Dd, Nn, Hh, Bc * Hh, 0, 1.0f, nullptr, nullptr);

    // ---- output gating + projection + cond gate + residual
    {
        long long n = (long long)BN * CS;
        gate_kernel<<<(int)((n + 255) / 256), 256, 0, stream>>>(upd, OGf, updg, n);
    }
    launch_gemm(stream, updg, CS, 0, 0, g_wout, CS, 1, 0, 0,
                out2, nullptr, CS, 0, 0, BN, CS, CS, 1, 1, 0, 1.0f, nullptr, nullptr);
    {
        long long n = (long long)BN * CS;
        cgate_res_kernel<<<(int)((n + 255) / 256), 256, 0, stream>>>(x, out2, CGl, b_cg, xa, n);
    }

    // ---- transition: LN -> SwiGLU -> proj -> masked residual
    ln_kernel<<<BN, 256, 0, stream>>>(xa, ffn_s, ffn_b, xlb, nullptr, CS);
    launch_gemm(stream, xlb, CS, 0, 0, g_wa, 2 * CS, 1, 0, 0,
                A1, nullptr, 2 * CS, 0, 0, BN, 2 * CS, CS, 1, 1, 0, 1.0f, nullptr, nullptr);
    launch_gemm(stream, xlb, CS, 0, 0, g_wb2, 2 * CS, 1, 0, 0,
                A2, nullptr, 2 * CS, 0, 0, BN, 2 * CS, CS, 1, 1, 0, 1.0f, nullptr, nullptr);
    {
        long long n = (long long)BN * 2 * CS;
        swiglu_kernel<<<(int)((n + 255) / 256), 256, 0, stream>>>(A1, A2, hg, n);
    }
    launch_gemm(stream, hg, 2 * CS, 0, 0, g_wo, CS, 1, 0, 0,
                Ff, nullptr, CS, 0, 0, BN, CS, 2 * CS, 1, 1, 0, 1.0f, nullptr, nullptr);
    {
        long long n = (long long)BN * CS;
        final_kernel<<<(int)((n + 255) / 256), 256, 0, stream>>>(xa, Ff, x_mask, out, n);
    }
}
